// GlobalMultiHeadAttention_41523743818190
// MI455X (gfx1250) — matrix operand order repaired and checked by reading, not compiled, not run on hardware
//
#include <hip/hip_runtime.h>

// ---------- types ----------
typedef __attribute__((ext_vector_type(16))) __bf16          v16bf;
typedef __attribute__((ext_vector_type(8)))  float           v8f;
typedef __attribute__((ext_vector_type(16))) unsigned short  usv16;
typedef __attribute__((ext_vector_type(8)))  unsigned short  usv8;
typedef __attribute__((ext_vector_type(4)))  float           f32x4;

// ---------- helpers ----------
__device__ __forceinline__ void split_bf16(float v, unsigned short &h, unsigned short &l) {
  unsigned int bits = __float_as_uint(v);
  h = (unsigned short)(bits >> 16);                       // truncate to bf16 (hi)
  float lf = v - __uint_as_float((unsigned int)h << 16);  // exact residual
  l = (unsigned short)(__float_as_uint(lf) >> 16);        // bf16 of residual (lo)
}

// 3-term compensated bf16 WMMA: (ah+al)(bh+bl) ~= ah*bh + ah*bl + al*bh  (drop al*bl ~ 2^-16)
__device__ __forceinline__ v8f wmma3(v16bf ah, v16bf al, v16bf bh, v16bf bl, v8f c) {
  c = __builtin_amdgcn_wmma_f32_16x16x32_bf16(false, ah, false, bh, (short)0, c, false, false);
  c = __builtin_amdgcn_wmma_f32_16x16x32_bf16(false, ah, false, bl, (short)0, c, false, false);
  c = __builtin_amdgcn_wmma_f32_16x16x32_bf16(false, al, false, bh, (short)0, c, false, false);
  return c;
}

// A operand (16x32 bf16) from global f32 x, row-major pitch 256.
// Lane layout: lanes 0-15 row=lane, K = {0..7,16..23}; lanes 16-31 row=lane-16, K = {8..15,24..31}.
// Caller passes ka = k0 + (half?8:0); we fetch [ka..ka+7] and [ka+16..ka+23].
__device__ __forceinline__ void load_a_x(const float* __restrict__ x, int N, int m, int ka,
                                         v16bf &ah, v16bf &al) {
  usv16 h = {}, l = {};
  if (m < N) {
    const f32x4* p = reinterpret_cast<const f32x4*>(x + (size_t)m * 256 + ka);
    f32x4 q[4] = { p[0], p[1], p[4], p[5] };
#pragma unroll
    for (int e = 0; e < 16; ++e) {
      unsigned short hh, ll;
      split_bf16(q[e >> 2][e & 3], hh, ll);
      h[e] = hh; l[e] = ll;
    }
  }
  ah = __builtin_bit_cast(v16bf, h);
  al = __builtin_bit_cast(v16bf, l);
}

// B operand (32x16 bf16) from pre-transposed weight Bt[n][k] (pitch 256, bf16 hi/lo planes).
// Lane layout: lanes 0-15 col=lane K={0..7,16..23}, lanes 16-31 col=lane-16 K={8..15,24..31} (kb = k0 + half*8; fetch [kb..kb+7],[kb+16..kb+23]).
__device__ __forceinline__ void load_b(const unsigned short* __restrict__ bthi,
                                       const unsigned short* __restrict__ btlo,
                                       int n, int kb, v16bf &bh, v16bf &bl) {
  usv8 h0 = *reinterpret_cast<const usv8*>(bthi + n * 256 + kb);
  usv8 h1 = *reinterpret_cast<const usv8*>(bthi + n * 256 + kb + 16);
  usv8 l0 = *reinterpret_cast<const usv8*>(btlo + n * 256 + kb);
  usv8 l1 = *reinterpret_cast<const usv8*>(btlo + n * 256 + kb + 16);
  bh = __builtin_bit_cast(v16bf, __builtin_shufflevector(h0, h1, 0,1,2,3,4,5,6,7,8,9,10,11,12,13,14,15));
  bl = __builtin_bit_cast(v16bf, __builtin_shufflevector(l0, l1, 0,1,2,3,4,5,6,7,8,9,10,11,12,13,14,15));
}

// A operand from an LDS bf16 plane laid out row-major [rows][256].
__device__ __forceinline__ v16bf load_a_lds(const unsigned short* hb, int row, int ka) {
  usv8 s0 = *reinterpret_cast<const usv8*>(hb + row * 256 + ka);
  usv8 s1 = *reinterpret_cast<const usv8*>(hb + row * 256 + ka + 16);
  return __builtin_bit_cast(v16bf,
      __builtin_shufflevector(s0, s1, 0,1,2,3,4,5,6,7,8,9,10,11,12,13,14,15));
}

__device__ __forceinline__ int lower_bound(const int* __restrict__ b, int n, int key) {
  int lo = 0, hi = n;
  while (lo < hi) { int mid = (lo + hi) >> 1; if (b[mid] < key) lo = mid + 1; else hi = mid; }
  return lo;
}

// ---------- kernel 0: transpose + hi/lo split weights ----------
// src: [256 x Ncols] f32 row-major.  dst planes: [Npad x 256] ushort, dst[n*256+k] = bf16(src[k][n]).
__global__ void conv_w(const float* __restrict__ src, unsigned short* __restrict__ hi,
                       unsigned short* __restrict__ lo, int Ncols, int Npad) {
  int idx = blockIdx.x * 256 + threadIdx.x;
  if (idx >= Npad * 256) return;
  int n = idx >> 8, k = idx & 255;
  float v = (n < Ncols) ? src[k * Ncols + n] : 0.f;
  unsigned short h, l; split_bf16(v, h, l);
  hi[idx] = h; lo[idx] = l;
}

// ---------- kernel 1: alpha = relu(x@Wg1) @ Wg2  -> [N,8] ----------
__global__ __launch_bounds__(128) void k_alpha(const float* __restrict__ x,
    const unsigned short* __restrict__ g1hi, const unsigned short* __restrict__ g1lo,
    const unsigned short* __restrict__ g2hi, const unsigned short* __restrict__ g2lo,
    float* __restrict__ alpha, int N) {
  __shared__ unsigned short sh_hi[32 * 256];
  __shared__ unsigned short sh_lo[32 * 256];
  const int tid = threadIdx.x;
  const int wave = tid >> 5, lane = tid & 31, half = lane >> 4, r = lane & 15;
  const int aoff = half * 8, kbo = half * 8;
  const int m0 = blockIdx.x * 32;

  // GEMM1: h = relu(x @ Wg1); each wave owns 64 cols, 2 M-tiles of 16 rows.
  v8f acc[2][4] = {};
  for (int k0 = 0; k0 < 256; k0 += 32) {
    v16bf ah[2], al[2];
#pragma unroll
    for (int mt = 0; mt < 2; ++mt)
      load_a_x(x, N, m0 + mt * 16 + r, k0 + aoff, ah[mt], al[mt]);
#pragma unroll
    for (int t = 0; t < 4; ++t) {
      v16bf bh, bl;
      load_b(g1hi, g1lo, wave * 64 + t * 16 + r, k0 + kbo, bh, bl);
#pragma unroll
      for (int mt = 0; mt < 2; ++mt) acc[mt][t] = wmma3(ah[mt], al[mt], bh, bl, acc[mt][t]);
    }
  }
  // relu + split -> LDS (C layout: lane=col, VGPR v -> row = half*8+v)
#pragma unroll
  for (int mt = 0; mt < 2; ++mt)
#pragma unroll
    for (int t = 0; t < 4; ++t) {
      int col = wave * 64 + t * 16 + r;
#pragma unroll
      for (int v = 0; v < 8; ++v) {
        int row = mt * 16 + half * 8 + v;
        unsigned short hh, ll;
        split_bf16(fmaxf(acc[mt][t][v], 0.f), hh, ll);
        sh_hi[row * 256 + col] = hh;
        sh_lo[row * 256 + col] = ll;
      }
    }
  __syncthreads();
  // GEMM2: alpha = h @ Wg2 (padded to 16 cols); waves 0/1 handle the two M-tiles.
  if (wave < 2) {
    const int mt = wave;
    v8f c = {};
    for (int k0 = 0; k0 < 256; k0 += 32) {
      v16bf ah = load_a_lds(sh_hi, mt * 16 + r, k0 + aoff);
      v16bf al = load_a_lds(sh_lo, mt * 16 + r, k0 + aoff);
      v16bf bh, bl;
      load_b(g2hi, g2lo, r, k0 + kbo, bh, bl);
      c = wmma3(ah, al, bh, bl, c);
    }
    if (r < 8) {
#pragma unroll
      for (int v = 0; v < 8; ++v) {
        int node = m0 + mt * 16 + half * 8 + v;
        if (node < N) alpha[node * 8 + r] = c[v];
      }
    }
  }
}

// ---------- kernel 2: per-segment max & sum(exp) per head ----------
__global__ __launch_bounds__(128) void k_stats(const int* __restrict__ batch,
    const float* __restrict__ alpha, float* __restrict__ mstat, float* __restrict__ sstat, int N) {
  __shared__ float red[16][8];
  const int b = blockIdx.x;
  const int start = lower_bound(batch, N, b);
  const int end   = lower_bound(batch, N, b + 1);
  const int tid = threadIdx.x, head = tid & 7, t = tid >> 3;   // 16 strides x 8 heads
  float mx = -INFINITY;
  for (int n = start + t; n < end; n += 16) mx = fmaxf(mx, alpha[n * 8 + head]);
  red[t][head] = mx;
  __syncthreads();
  if (t == 0) { float m = red[0][head];
    for (int i = 1; i < 16; ++i) m = fmaxf(m, red[i][head]); red[0][head] = m; }
  __syncthreads();
  const float m = red[0][head];
  __syncthreads();
  float sm = 0.f;
  for (int n = start + t; n < end; n += 16) sm += __expf(alpha[n * 8 + head] - m);
  red[t][head] = sm;
  __syncthreads();
  if (t == 0) { float s = 0.f;
    for (int i = 0; i < 16; ++i) s += red[i][head];
    mstat[b * 8 + head] = m; sstat[b * 8 + head] = s; }
}

// ---------- kernel 3: per-segment fused MLP + gated pooling ----------
__global__ __launch_bounds__(128) void k_pool(const float* __restrict__ x,
    const int* __restrict__ batch,
    const unsigned short* __restrict__ n1hi, const unsigned short* __restrict__ n1lo,
    const unsigned short* __restrict__ n2hi, const unsigned short* __restrict__ n2lo,
    const float* __restrict__ b1, const float* __restrict__ b2,
    const float* __restrict__ alpha, const float* __restrict__ mstat,
    const float* __restrict__ sstat, float* __restrict__ out, int N) {
  __shared__ unsigned short sh_hi[32 * 256];
  __shared__ unsigned short sh_lo[32 * 256];
  __shared__ float sgate[32][8];
  __shared__ float spool[4][2][64];
  const int b = blockIdx.x;
  const int start = lower_bound(batch, N, b);
  const int end   = lower_bound(batch, N, b + 1);
  const int tid = threadIdx.x;
  const int wave = tid >> 5, lane = tid & 31, half = lane >> 4, r = lane & 15;
  const int aoff = half * 8, kbo = half * 8;
  const int gh = tid & 7, gn = tid >> 3;
  const float mh    = mstat[b * 8 + gh];
  const float inv_s = 1.f / (sstat[b * 8 + gh] + 1e-16f);
  float b1v[4], b2v[4], pacc[4] = {0.f, 0.f, 0.f, 0.f};
#pragma unroll
  for (int t = 0; t < 4; ++t) {
    int col = wave * 64 + t * 16 + r;
    b1v[t] = b1[col]; b2v[t] = b2[col];
  }
  const int headc = r & 7;  // (wave*64 + t*16 + r) % 8 == r % 8

  for (int c0 = start; c0 < end; c0 += 32) {
    // gates for this 32-node chunk (zero outside the segment)
#pragma unroll
    for (int gg = 0; gg < 2; ++gg) {
      int nn = gn + gg * 16, node = c0 + nn;
      float g = 0.f;
      if (node < end) g = __expf(alpha[node * 8 + gh] - mh) * inv_s;
      sgate[nn][gh] = g;
    }
    // GEMM1: h = relu(x @ Wn1 + b1)
    v8f acc[2][4] = {};
    for (int k0 = 0; k0 < 256; k0 += 32) {
      v16bf ah[2], al[2];
#pragma unroll
      for (int mt = 0; mt < 2; ++mt)
        load_a_x(x, N, c0 + mt * 16 + r, k0 + aoff, ah[mt], al[mt]);
#pragma unroll
      for (int t = 0; t < 4; ++t) {
        v16bf bh, bl;
        load_b(n1hi, n1lo, wave * 64 + t * 16 + r, k0 + kbo, bh, bl);
#pragma unroll
        for (int mt = 0; mt < 2; ++mt) acc[mt][t] = wmma3(ah[mt], al[mt], bh, bl, acc[mt][t]);
      }
    }
#pragma unroll
    for (int mt = 0; mt < 2; ++mt)
#pragma unroll
      for (int t = 0; t < 4; ++t) {
        int col = wave * 64 + t * 16 + r;
#pragma unroll
        for (int v = 0; v < 8; ++v) {
          int row = mt * 16 + half * 8 + v;
          unsigned short hh, ll;
          split_bf16(fmaxf(acc[mt][t][v] + b1v[t], 0.f), hh, ll);
          sh_hi[row * 256 + col] = hh;
          sh_lo[row * 256 + col] = ll;
        }
      }
    __syncthreads();
    // GEMM2: f = h @ Wn2 (+b2), then gated accumulation
    v8f fac[2][4] = {};
    for (int k0 = 0; k0 < 256; k0 += 32) {
      v16bf ah[2], al[2];
#pragma unroll
      for (int mt = 0; mt < 2; ++mt) {
        ah[mt] = load_a_lds(sh_hi, mt * 16 + r, k0 + aoff);
        al[mt] = load_a_lds(sh_lo, mt * 16 + r, k0 + aoff);
      }
#pragma unroll
      for (int t = 0; t < 4; ++t) {
        v16bf bh, bl;
        load_b(n2hi, n2lo, wave * 64 + t * 16 + r, k0 + kbo, bh, bl);
#pragma unroll
        for (int mt = 0; mt < 2; ++mt) fac[mt][t] = wmma3(ah[mt], al[mt], bh, bl, fac[mt][t]);
      }
    }
#pragma unroll
    for (int mt = 0; mt < 2; ++mt)
#pragma unroll
      for (int v = 0; v < 8; ++v) {
        int row = mt * 16 + half * 8 + v;
        float g = sgate[row][headc];
#pragma unroll
        for (int t = 0; t < 4; ++t) pacc[t] += g * (fac[mt][t][v] + b2v[t]);
      }
    __syncthreads();
  }
  // combine lane halves (rows 0-7 vs 8-15 of each tile live in lane pairs r / r+16)
#pragma unroll
  for (int t = 0; t < 4; ++t) spool[wave][half][t * 16 + r] = pacc[t];
  __syncthreads();
  for (int c = tid; c < 256; c += 128) {
    int w = c >> 6, tt = (c >> 4) & 3, rr = c & 15;
    out[(size_t)b * 256 + c] = spool[w][0][tt * 16 + rr] + spool[w][1][tt * 16 + rr];
  }
}

// ---------- host launcher ----------
extern "C" void kernel_launch(void* const* d_in, const int* in_sizes, int n_in,
                              void* d_out, int out_size, void* d_ws, size_t ws_size,
                              hipStream_t stream) {
  const float* x   = (const float*)d_in[0];
  const int*   bat = (const int*)d_in[1];
  const float* Wg1 = (const float*)d_in[2];
  const float* Wg2 = (const float*)d_in[3];
  const float* Wn1 = (const float*)d_in[4];
  const float* b1  = (const float*)d_in[5];
  const float* Wn2 = (const float*)d_in[6];
  const float* b2  = (const float*)d_in[7];
  float* out = (float*)d_out;
  const int N = in_sizes[0] / 256;
  const int B = out_size / 256;

  char* ws = (char*)d_ws;
  auto alloc = [&](size_t bytes) { char* p = ws; ws += (bytes + 255) & ~(size_t)255; return p; };
  float* alpha = (float*)alloc((size_t)N * 8 * sizeof(float));
  float* mstat = (float*)alloc((size_t)B * 8 * sizeof(float));
  float* sstat = (float*)alloc((size_t)B * 8 * sizeof(float));
  unsigned short* g1hi = (unsigned short*)alloc(256 * 256 * 2);
  unsigned short* g1lo = (unsigned short*)alloc(256 * 256 * 2);
  unsigned short* n1hi = (unsigned short*)alloc(256 * 256 * 2);
  unsigned short* n1lo = (unsigned short*)alloc(256 * 256 * 2);
  unsigned short* n2hi = (unsigned short*)alloc(256 * 256 * 2);
  unsigned short* n2lo = (unsigned short*)alloc(256 * 256 * 2);
  unsigned short* g2hi = (unsigned short*)alloc(16 * 256 * 2);
  unsigned short* g2lo = (unsigned short*)alloc(16 * 256 * 2);

  conv_w<<<256, 256, 0, stream>>>(Wg1, g1hi, g1lo, 256, 256);
  conv_w<<<256, 256, 0, stream>>>(Wn1, n1hi, n1lo, 256, 256);
  conv_w<<<256, 256, 0, stream>>>(Wn2, n2hi, n2lo, 256, 256);
  conv_w<<<16,  256, 0, stream>>>(Wg2, g2hi, g2lo, 8, 16);

  k_alpha<<<(N + 31) / 32, 128, 0, stream>>>(x, g1hi, g1lo, g2hi, g2lo, alpha, N);
  k_stats<<<B, 128, 0, stream>>>(bat, alpha, mstat, sstat, N);
  k_pool <<<B, 128, 0, stream>>>(x, bat, n1hi, n1lo, n2hi, n2lo, b1, b2,
                                 alpha, mstat, sstat, out, N);
}
